// MultiHeadAttention_38293928411638
// MI455X (gfx1250) — matrix operand order repaired and checked by reading, not compiled, not run on hardware
//
#include <hip/hip_runtime.h>
#include <math.h>

typedef __bf16 bf16_t;
typedef bf16_t v16bf __attribute__((ext_vector_type(16)));
typedef bf16_t v8bf  __attribute__((ext_vector_type(8)));
typedef float  v8f   __attribute__((ext_vector_type(8)));

constexpr int kBatch = 2;
constexpr int kS   = 2048;
constexpr int kD   = 2048;
constexpr int kHQ  = 32;
constexpr int kHKV = 8;
constexpr int kDH  = 64;
constexpr int kQD  = kHQ * kDH;   // 2048
constexpr int kKD  = kHKV * kDH;  // 512
constexpr int kM   = kBatch * kS; // 4096

__device__ __forceinline__ bf16_t f2bf(float f) {
  union { float f; unsigned int u; } v; v.f = f;
  unsigned int r = v.u + 0x7FFFu + ((v.u >> 16) & 1u);  // round-to-nearest-even
  unsigned short h = (unsigned short)(r >> 16);
  return __builtin_bit_cast(bf16_t, h);
}

// Combine two contiguous 8-element (16B) vector loads into a v16bf.
__device__ __forceinline__ v16bf load16(const bf16_t* p0, const bf16_t* p1) {
  v8bf a = *(const v8bf*)p0;
  v8bf b = *(const v8bf*)p1;
  return __builtin_shufflevector(a, b, 0, 1, 2, 3, 4, 5, 6, 7,
                                 8, 9, 10, 11, 12, 13, 14, 15);
}

// ---------------------------------------------------------------- conversion
__global__ void __launch_bounds__(256)
cvt_f32_bf16(const float* __restrict__ in, bf16_t* __restrict__ out, long n) {
  long i = (long)blockIdx.x * blockDim.x + threadIdx.x;
  if (i < n) out[i] = f2bf(in[i]);
}

// Transpose + convert: in [R][C] f32 -> out [C][R] bf16. 32x32 LDS tiles.
__global__ void __launch_bounds__(256)
cvt_transpose_bf16(const float* __restrict__ in, bf16_t* __restrict__ out,
                   int R, int C) {
  __shared__ float tile[32][33];
  const int bx = blockIdx.x % (C / 32);  // column tile
  const int by = blockIdx.x / (C / 32);  // row tile
  const int tx = threadIdx.x & 31;
  const int ty = threadIdx.x >> 5;       // 0..7
#pragma unroll
  for (int i = 0; i < 32; i += 8)
    tile[ty + i][tx] = in[(long)(by * 32 + ty + i) * C + bx * 32 + tx];
  __syncthreads();
#pragma unroll
  for (int i = 0; i < 32; i += 8)
    out[(long)(bx * 32 + ty + i) * R + by * 32 + tx] = f2bf(tile[tx][ty + i]);
}

// ---------------------------------------------------------------- GEMM
// C[M,N] = A[M,K] * B[K,N], A row-major [M][K], B given TRANSPOSED [N][K].
// One wave computes a 16(M) x 64(N) tile; K-loop step 32, double-buffered so
// WMMAs consume tiles loaded one iteration earlier (latency overlap).
__global__ void __launch_bounds__(256)
gemm_bf16(const bf16_t* __restrict__ A, const bf16_t* __restrict__ Bt,
          float* __restrict__ Cf, bf16_t* __restrict__ Cb, int storeT,
          int M, int N, int Kd) {
  const int lane   = threadIdx.x & 31;
  const int w      = threadIdx.x >> 5;
  const int waveId = blockIdx.x * 8 + w;
  const int ntiles = N >> 6;
  const int row0   = (waveId / ntiles) << 4;
  const int n0     = (waveId % ntiles) << 6;
  const int hf     = lane >> 4;
  const int ln     = lane & 15;

  v8f acc0 = {}, acc1 = {}, acc2 = {}, acc3 = {};
  const bf16_t* pa_row = A + (long)(row0 + ln) * Kd + 16 * hf;
  const bf16_t* pb0 = Bt + (long)(n0 + 0  + ln) * Kd + 16 * hf;
  const bf16_t* pb1 = Bt + (long)(n0 + 16 + ln) * Kd + 16 * hf;
  const bf16_t* pb2 = Bt + (long)(n0 + 32 + ln) * Kd + 16 * hf;
  const bf16_t* pb3 = Bt + (long)(n0 + 48 + ln) * Kd + 16 * hf;

  // prologue loads (k0 = 0)
  v16bf a  = load16(pa_row, pa_row + 8);
  v16bf b0 = load16(pb0, pb0 + 8);
  v16bf b1 = load16(pb1, pb1 + 8);
  v16bf b2 = load16(pb2, pb2 + 8);
  v16bf b3 = load16(pb3, pb3 + 8);

  for (int k0 = 0; k0 < Kd - 32; k0 += 32) {
    const int kn = k0 + 32;
    // issue next-iteration loads before consuming current tiles
    v16bf an  = load16(pa_row + kn, pa_row + kn + 8);
    v16bf bn0 = load16(pb0 + kn, pb0 + kn + 8);
    v16bf bn1 = load16(pb1 + kn, pb1 + kn + 8);
    v16bf bn2 = load16(pb2 + kn, pb2 + kn + 8);
    v16bf bn3 = load16(pb3 + kn, pb3 + kn + 8);
    __builtin_prefetch(pa_row + kn + 32, 0, 1);  // speculative, safe past end
    acc0 = __builtin_amdgcn_wmma_f32_16x16x32_bf16(false, a, false, b0, (short)0, acc0, false, false);
    acc1 = __builtin_amdgcn_wmma_f32_16x16x32_bf16(false, a, false, b1, (short)0, acc1, false, false);
    acc2 = __builtin_amdgcn_wmma_f32_16x16x32_bf16(false, a, false, b2, (short)0, acc2, false, false);
    acc3 = __builtin_amdgcn_wmma_f32_16x16x32_bf16(false, a, false, b3, (short)0, acc3, false, false);
    a = an; b0 = bn0; b1 = bn1; b2 = bn2; b3 = bn3;
  }
  acc0 = __builtin_amdgcn_wmma_f32_16x16x32_bf16(false, a, false, b0, (short)0, acc0, false, false);
  acc1 = __builtin_amdgcn_wmma_f32_16x16x32_bf16(false, a, false, b1, (short)0, acc1, false, false);
  acc2 = __builtin_amdgcn_wmma_f32_16x16x32_bf16(false, a, false, b2, (short)0, acc2, false, false);
  acc3 = __builtin_amdgcn_wmma_f32_16x16x32_bf16(false, a, false, b3, (short)0, acc3, false, false);

  if (Cf) {
#pragma unroll
    for (int r = 0; r < 8; ++r) {
      float* pc = Cf + (long)(row0 + r + 8 * hf) * N + n0 + ln;
      pc[0] = acc0[r]; pc[16] = acc1[r]; pc[32] = acc2[r]; pc[48] = acc3[r];
    }
  } else if (storeT) {
    // store transposed: Cb[n][m], each lane writes 8 consecutive m as one 16B store
#pragma unroll
    for (int nt = 0; nt < 4; ++nt) {
      const v8f& cc = (nt == 0) ? acc0 : (nt == 1) ? acc1 : (nt == 2) ? acc2 : acc3;
      v8bf o;
#pragma unroll
      for (int r = 0; r < 8; ++r) o[r] = f2bf(cc[r]);
      *(v8bf*)(Cb + (long)(n0 + nt * 16 + ln) * M + row0 + 8 * hf) = o;
    }
  } else {
#pragma unroll
    for (int r = 0; r < 8; ++r) {
      bf16_t* pc = Cb + (long)(row0 + r + 8 * hf) * N + n0 + ln;
      pc[0]  = f2bf(acc0[r]); pc[16] = f2bf(acc1[r]);
      pc[32] = f2bf(acc2[r]); pc[48] = f2bf(acc3[r]);
    }
  }
}

// ---------------------------------------------------------------- attention
struct KTiles { v16bf t00, t01, t10, t11; };

__device__ __forceinline__ KTiles loadK(const bf16_t* Kbase, int k0, int ln, int hf) {
  const bf16_t* p0 = Kbase + (long)(k0 + ln) * kKD + 16 * hf;
  const bf16_t* p1 = p0 + (long)16 * kKD;
  KTiles t;
  t.t00 = load16(p0, p0 + 8);        // score tile 0, d in [0,32)
  t.t01 = load16(p0 + 32, p0 + 40);  // score tile 0, d in [32,64)
  t.t10 = load16(p1, p1 + 8);        // score tile 1, d in [0,32)
  t.t11 = load16(p1 + 32, p1 + 40);  // score tile 1, d in [32,64)
  return t;
}

// One wave per (batch, q-head, 16-row q block). Flash-style online softmax.
// Q: [B,S,HQ,DH] bf16, K: [B,S,HKV,DH] bf16, Vt: [HKV*DH][B*S] bf16 (transposed),
// Ctx out: [B,S,HQ,DH] bf16.
__global__ void __launch_bounds__(128)
flash_attn(const bf16_t* __restrict__ Q, const bf16_t* __restrict__ K,
           const bf16_t* __restrict__ Vt, bf16_t* __restrict__ Ctx) {
  __shared__ bf16_t pbuf[4][16][32];  // per-wave 16x32 P staging

  const int lane   = threadIdx.x & 31;
  const int w      = threadIdx.x >> 5;
  const int waveId = blockIdx.x * 4 + w;
  const int qb = waveId & 127;          // S/16 = 128 q-blocks
  const int h  = (waveId >> 7) & 31;    // HQ = 32
  const int bb = waveId >> 12;          // batch
  const int hk = h >> 2;                // NUM_REP = 4
  const int q0 = qb << 4;
  const int hf = lane >> 4;
  const int ln = lane & 15;

  const bf16_t* Qbase = Q + (long)bb * kS * kQD + h * kDH;
  const bf16_t* Kbase = K + (long)bb * kS * kKD + hk * kDH;

  // Q A-tiles for d in [0,32) and [32,64)
  const bf16_t* qp = Qbase + (long)(q0 + ln) * kQD + 16 * hf;
  v16bf a0 = load16(qp, qp + 8);
  v16bf a1 = load16(qp + 32, qp + 40);

  // Vt row pointers: lane owns dim d = nt*16+ln, reads contiguous kv positions
  const bf16_t* pv0 = Vt + (long)(hk * kDH + 0  + ln) * kM + bb * kS + 16 * hf;
  const bf16_t* pv1 = Vt + (long)(hk * kDH + 16 + ln) * kM + bb * kS + 16 * hf;
  const bf16_t* pv2 = Vt + (long)(hk * kDH + 32 + ln) * kM + bb * kS + 16 * hf;
  const bf16_t* pv3 = Vt + (long)(hk * kDH + 48 + ln) * kM + bb * kS + 16 * hf;

  v8f ctx0 = {}, ctx1 = {}, ctx2 = {}, ctx3 = {};
  float mrow[8], lrow[8];
#pragma unroll
  for (int r = 0; r < 8; ++r) { mrow[r] = -INFINITY; lrow[r] = 0.f; }

  // prologue: K tiles for chunk 0
  KTiles kb = loadK(Kbase, 0, ln, hf);

  for (int j = 0; j < kS / 32; ++j) {
    const int k0 = j * 32;
    // ---- V tiles for this chunk: issued early, consumed after softmax
    v16bf vb0 = load16(pv0 + k0, pv0 + k0 + 8);
    v16bf vb1 = load16(pv1 + k0, pv1 + k0 + 8);
    v16bf vb2 = load16(pv2 + k0, pv2 + k0 + 8);
    v16bf vb3 = load16(pv3 + k0, pv3 + k0 + 8);
    // ---- scores from K tiles loaded one iteration earlier
    v8f s0 = {}, s1 = {};
    s0 = __builtin_amdgcn_wmma_f32_16x16x32_bf16(false, a0, false, kb.t00, (short)0, s0, false, false);
    s0 = __builtin_amdgcn_wmma_f32_16x16x32_bf16(false, a1, false, kb.t01, (short)0, s0, false, false);
    s1 = __builtin_amdgcn_wmma_f32_16x16x32_bf16(false, a0, false, kb.t10, (short)0, s1, false, false);
    s1 = __builtin_amdgcn_wmma_f32_16x16x32_bf16(false, a1, false, kb.t11, (short)0, s1, false, false);
    // ---- prefetch next chunk's K (branchless clamp on last iteration)
    const int kn = (j + 1 < kS / 32) ? k0 + 32 : k0;
    KTiles kbn = loadK(Kbase, kn, ln, hf);
    // ---- online softmax (row r+8*hf lives in vgpr index r, reduce over 16-lane half)
#pragma unroll
    for (int r = 0; r < 8; ++r) {
      float x0 = s0[r] * 0.125f, x1 = s1[r] * 0.125f;  // 1/sqrt(64)
      float cm = fmaxf(x0, x1);
#pragma unroll
      for (int off = 1; off < 16; off <<= 1) cm = fmaxf(cm, __shfl_xor(cm, off, 32));
      float mnew  = fmaxf(mrow[r], cm);
      float alpha = __expf(mrow[r] - mnew);
      float p0 = __expf(x0 - mnew), p1 = __expf(x1 - mnew);
      float rs = p0 + p1;
#pragma unroll
      for (int off = 1; off < 16; off <<= 1) rs += __shfl_xor(rs, off, 32);
      lrow[r] = lrow[r] * alpha + rs;
      mrow[r] = mnew;
      ctx0[r] *= alpha; ctx1[r] *= alpha; ctx2[r] *= alpha; ctx3[r] *= alpha;
      const int m = r + 8 * hf;
      pbuf[w][m][ln]      = f2bf(p0);
      pbuf[w][m][16 + ln] = f2bf(p1);
    }
    __syncthreads();
    // ---- reload P in A-matrix layout (two 16B LDS reads per lane)
    const bf16_t* pp = &pbuf[w][ln][16 * hf];
    v16bf pA = load16(pp, pp + 8);
    // ---- P (16x32) x V (32x64) accumulated into 4 C tiles
    ctx0 = __builtin_amdgcn_wmma_f32_16x16x32_bf16(false, pA, false, vb0, (short)0, ctx0, false, false);
    ctx1 = __builtin_amdgcn_wmma_f32_16x16x32_bf16(false, pA, false, vb1, (short)0, ctx1, false, false);
    ctx2 = __builtin_amdgcn_wmma_f32_16x16x32_bf16(false, pA, false, vb2, (short)0, ctx2, false, false);
    ctx3 = __builtin_amdgcn_wmma_f32_16x16x32_bf16(false, pA, false, vb3, (short)0, ctx3, false, false);
    __syncthreads();
    kb = kbn;
  }

  // ---- epilogue: normalize and store context (bf16, [B,S,HQ,DH])
  bf16_t* Co = Ctx + (long)(bb * kS + q0) * kQD + h * kDH;
#pragma unroll
  for (int r = 0; r < 8; ++r) {
    float inv = 1.0f / lrow[r];
    bf16_t* co = Co + (long)(r + 8 * hf) * kQD + ln;
    co[0]  = f2bf(ctx0[r] * inv);
    co[16] = f2bf(ctx1[r] * inv);
    co[32] = f2bf(ctx2[r] * inv);
    co[48] = f2bf(ctx3[r] * inv);
  }
}

// ---------------------------------------------------------------- launch
extern "C" void kernel_launch(void* const* d_in, const int* in_sizes, int n_in,
                              void* d_out, int out_size, void* d_ws, size_t ws_size,
                              hipStream_t stream) {
  const float* x  = (const float*)d_in[0];
  const float* Wq = (const float*)d_in[1];
  const float* Wk = (const float*)d_in[2];
  const float* Wv = (const float*)d_in[3];
  const float* Wo = (const float*)d_in[4];
  float* out = (float*)d_out;

  const long nX  = (long)kM * kD;
  const long nWq = (long)kD * kQD;
  const long nWk = (long)kD * kKD;
  const long nWo = (long)kQD * kD;
  const long nQ  = (long)kM * kQD;
  const long nK  = (long)kM * kKD;

  char* ws = (char*)d_ws;
  size_t off = 0;
  auto alloc = [&](long elems) -> bf16_t* {
    bf16_t* p = (bf16_t*)(ws + off);
    off += ((size_t)elems * sizeof(bf16_t) + 255) & ~(size_t)255;
    return p;
  };
  bf16_t* xb   = alloc(nX);   // x            [M][D]       bf16
  bf16_t* WqT  = alloc(nWq);  // Wq^T         [QD][D]      bf16
  bf16_t* WkT  = alloc(nWk);  // Wk^T         [KD][D]      bf16
  bf16_t* WvT  = alloc(nWk);  // Wv^T         [KD][D]      bf16
  bf16_t* WoT  = alloc(nWo);  // Wo^T         [D][QD]      bf16
  bf16_t* Qb   = alloc(nQ);   // Q            [M][QD]      bf16
  bf16_t* Kb   = alloc(nK);   // K            [M][KD]      bf16
  bf16_t* VbT  = alloc(nK);   // V^T          [KD][M]      bf16
  bf16_t* Cxb  = alloc(nQ);   // context      [M][QD]      bf16

  cvt_f32_bf16<<<(unsigned)((nX + 255) / 256), 256, 0, stream>>>(x, xb, nX);
  cvt_transpose_bf16<<<(kD / 32) * (kQD / 32), 256, 0, stream>>>(Wq, WqT, kD, kQD);
  cvt_transpose_bf16<<<(kD / 32) * (kKD / 32), 256, 0, stream>>>(Wk, WkT, kD, kKD);
  cvt_transpose_bf16<<<(kD / 32) * (kKD / 32), 256, 0, stream>>>(Wv, WvT, kD, kKD);
  cvt_transpose_bf16<<<(kQD / 32) * (kD / 32), 256, 0, stream>>>(Wo, WoT, kQD, kD);

  // Q/K projections (bf16 out, row-major); V projection stored transposed
  gemm_bf16<<<(kM / 16) * (kQD / 64) / 8, 256, 0, stream>>>(xb, WqT, nullptr, Qb, 0, kM, kQD, kD);
  gemm_bf16<<<(kM / 16) * (kKD / 64) / 8, 256, 0, stream>>>(xb, WkT, nullptr, Kb, 0, kM, kKD, kD);
  gemm_bf16<<<(kM / 16) * (kKD / 64) / 8, 256, 0, stream>>>(xb, WvT, nullptr, VbT, 1, kM, kKD, kD);

  // attention: B*HQ*(S/16) = 8192 waves, 4 waves/block
  flash_attn<<<kBatch * kHQ * (kS / 16) / 4, 128, 0, stream>>>(Qb, Kb, VbT, Cxb);

  // output projection (f32 out)
  gemm_bf16<<<(kM / 16) * (kD / 64) / 8, 256, 0, stream>>>(Cxb, WoT, out, nullptr, 0, kM, kD, kQD);
}
